// TRTModel_47837345743214
// MI455X (gfx1250) — matrix operand order repaired and checked by reading, not compiled, not run on hardware
//
#include <hip/hip_runtime.h>
#include <hip/hip_bf16.h>

typedef __attribute__((ext_vector_type(16))) _Float16 v16h;
typedef __attribute__((ext_vector_type(8)))  float    v8f;

union V16H { v16h v; _Float16 e[16]; };
union V8F  { v8f  v; float    e[8];  };

#define B_   2
#define N_   6
#define D_   70
#define C_   80
#define CP_  80
#define H_   32
#define W_   88
#define Y_   128
#define X_   128
#define CT_  160
#define BN_  12
#define HW_  (H_*W_)      // 2816
#define NT2_ (HW_/16)     // 176 N-tiles for k_project

// ---- workspace layout (bytes) ----
#define OFF_ACC   ((size_t)0)                       // B*Y*X*CT f32   = 20,971,520
#define OFF_CNT   ((size_t)20971520)                // B*Y*X   f32   =    131,072
#define OFF_DV    ((size_t)21102592)                // BN*W*D*H f16  =  4,730,880
#define OFF_VMAX  ((size_t)25833472)                // BN*W*H  f32   =    135,168
#define OFF_IMG   ((size_t)25968640)                // BN*C*H*W f16  =  5,406,720
#define OFF_WCAT  ((size_t)31375360)                // 160*96  f16   =     30,720
#define OFF_WAB   ((size_t)31406080)                // BN*W*CT*H f16 = 10,813,440

// ---------------------------------------------------------------------------
// Kernel 0: zero BEV accumulator/counts; pack Wcat = [w1;w2] as f16, K padded
// ---------------------------------------------------------------------------
__global__ void k_init(float* __restrict__ acc, float* __restrict__ cnt,
                       _Float16* __restrict__ wcat, const float* __restrict__ w_agg) {
  size_t gid = (size_t)blockIdx.x * blockDim.x + threadIdx.x;
  size_t nacc = (size_t)B_ * Y_ * X_ * CT_;
  if (gid < nacc) acc[gid] = 0.0f;
  if (gid < (size_t)B_ * Y_ * X_) cnt[gid] = 0.0f;
  if (gid < (size_t)CT_ * 96) {
    int oo = (int)(gid / 96), k = (int)(gid % 96);
    float v = 0.0f;
    if (k < C_)
      v = (oo < C_) ? w_agg[oo * (2 * C_) + k] : w_agg[(oo - C_) * (2 * C_) + C_ + k];
    wcat[gid] = (_Float16)v;
  }
}

// ---------------------------------------------------------------------------
// Kernel 1: per (n,h): softmax over D, *valid -> dv (f16, [n][w][d][h]);
//           valid_max ([n][w][h]); img_feat f16 copy ([n][c][h][w]).
// ---------------------------------------------------------------------------
__global__ void k_softmax(const float* __restrict__ df, const float* __restrict__ gv,
                          _Float16* __restrict__ dv, _Float16* __restrict__ img16,
                          float* __restrict__ vmax) {
  int blk = blockIdx.x;
  int n = blk / H_, h = blk % H_;
  int tid = threadIdx.x;
  const float* dfp = df + (size_t)n * (D_ + C_) * HW_ + h * W_;
  const float* gvp = gv + (size_t)n * D_ * HW_ + h * W_;
  for (int w = tid; w < W_; w += blockDim.x) {
    float m = -1e30f;
    for (int d = 0; d < D_; ++d) m = fmaxf(m, dfp[(size_t)d * HW_ + w]);
    float s = 0.0f;
    for (int d = 0; d < D_; ++d) s += __expf(dfp[(size_t)d * HW_ + w] - m);
    float inv = 1.0f / s;
    float vm = 0.0f;
    for (int d = 0; d < D_; ++d) {
      float vld = gvp[(size_t)d * HW_ + w];
      vm = fmaxf(vm, vld);
      float val = __expf(dfp[(size_t)d * HW_ + w] - m) * inv * vld;
      dv[(((size_t)n * W_ + w) * D_ + d) * H_ + h] = (_Float16)val;
    }
    vmax[((size_t)n * W_ + w) * H_ + h] = vm;
  }
  for (int t = tid; t < C_ * W_; t += blockDim.x) {
    int c = t / W_, w = t % W_;
    img16[(((size_t)n * C_ + c) * H_ + h) * W_ + w] =
        (_Float16)df[((size_t)n * (D_ + C_) + D_ + c) * HW_ + h * W_ + w];
  }
}

// ---------------------------------------------------------------------------
// Kernel 2: WAB[n] (160 x HW) = Wcat (160x80, K-pad 96) x img_feat (80 x HW)
// f16 WMMA 16x16x32, 3 K-chunks; B tile staged transposed in LDS so both
// fragment gathers are contiguous (ds_load_b128). Output layout [n][w][oo][h].
// ---------------------------------------------------------------------------
__global__ void k_project(const _Float16* __restrict__ img16,
                          const _Float16* __restrict__ wcat,
                          _Float16* __restrict__ wab) {
  __shared__ _Float16 lw[CT_][96];   // A: [oo][k], k padded
  __shared__ _Float16 lf[16][96];    // B^T: [col][c], c padded with zeros
  int blk = blockIdx.x;
  int n = blk / NT2_, nt = blk % NT2_;
  int tid = threadIdx.x;
  int hwb = nt * 16;
  const _Float16* F = img16 + (size_t)n * C_ * HW_;
  for (int t = tid; t < CT_ * 96; t += blockDim.x) lw[t / 96][t % 96] = wcat[t];
  for (int t = tid; t < 96 * 16; t += blockDim.x) {
    int c = t / 16, col = t % 16;                 // coalesced on global (col fastest)
    _Float16 v = (c < C_) ? F[(size_t)c * HW_ + hwb + col] : (_Float16)0.0f;
    lf[col][c] = v;
  }
  __syncthreads();
  int lane = tid & 31, wv = tid >> 5;
  int M = lane & 15, hi = lane >> 4;
  for (int mt = wv; mt < 10; mt += 8) {
    V8F acc;
#pragma unroll
    for (int r = 0; r < 8; ++r) acc.e[r] = 0.0f;
    for (int kc = 0; kc < 3; ++kc) {
      V16H a, b;
#pragma unroll
      for (int i = 0; i < 16; ++i) a.e[i] = lw[mt * 16 + M][kc * 32 + 16 * hi + i];
#pragma unroll
      for (int i = 0; i < 16; ++i) b.e[i] = lf[M][kc * 32 + 16 * hi + i];
      acc.v = __builtin_amdgcn_wmma_f32_16x16x32_f16(false, a.v, false, b.v,
                                                     (short)0, acc.v, false, false);
    }
    int hw = hwb + M, h = hw / W_, w = hw % W_;
#pragma unroll
    for (int r = 0; r < 8; ++r) {
      int oo = mt * 16 + r + 8 * hi;
      wab[(((size_t)n * W_ + w) * CT_ + oo) * H_ + h] = (_Float16)acc.e[r];
    }
  }
}

// ---------------------------------------------------------------------------
// Kernel 3: per (n,w): out[o,d] = sum_h WA[o,h]*dv[d,h]  (25 WMMAs, K=32)
//           + s[o]*radar[d] + bias[o];  then fused atomic scatter into BEV.
// ---------------------------------------------------------------------------
__global__ void k_context_scatter(const _Float16* __restrict__ wab,
                                  const _Float16* __restrict__ dv,
                                  const float* __restrict__ vmax,
                                  const float* __restrict__ radar,
                                  const float* __restrict__ b_agg,
                                  const float* __restrict__ pts,
                                  const int* __restrict__ geom,
                                  float* __restrict__ acc, float* __restrict__ cnt) {
  __shared__ _Float16 lA[C_][H_];   // WA rows (oo < 80)
  __shared__ _Float16 lB[C_][H_];   // WB rows (oo >= 80)
  __shared__ _Float16 lD[80][H_];   // dv, d padded to 80
  __shared__ float    lO[80][80];   // out[o][d]
  __shared__ float    sS[C_];
  __shared__ float    sV[H_];
  __shared__ int      sIdx[D_];
  int blk = blockIdx.x;
  int n = blk / W_, w = blk % W_;
  int tid = threadIdx.x;

  const _Float16* wabp = wab + ((size_t)n * W_ + w) * CT_ * H_;   // contiguous
  const _Float16* dvp  = dv  + ((size_t)n * W_ + w) * D_ * H_;    // contiguous
  const float*    vmp  = vmax + ((size_t)n * W_ + w) * H_;

  for (int t = tid; t < C_ * H_; t += blockDim.x) {
    lA[t / H_][t % H_] = wabp[t];
    lB[t / H_][t % H_] = wabp[C_ * H_ + t];
  }
  for (int t = tid; t < 80 * H_; t += blockDim.x)
    lD[t / H_][t % H_] = (t < D_ * H_) ? dvp[t] : (_Float16)0.0f;
  for (int t = tid; t < H_; t += blockDim.x) sV[t] = vmp[t];
  if (tid < D_) {
    size_t gbase = (((size_t)n * D_ + tid) * W_ + w) * 3;
    int gx = geom[gbase + 0], gy = geom[gbase + 1];
    sIdx[tid] = (gx >= 0 && gx < X_ && gy >= 0 && gy < Y_) ? gy * X_ + gx : -1;
  }
  // warm L2 for the strided pts_context reads used in the scatter phase
  for (int cp = tid; cp < CP_; cp += blockDim.x)
    __builtin_prefetch(&pts[(((size_t)n * CP_ + cp) * D_) * W_ + w], 0, 1);
  __syncthreads();

  int lane = tid & 31, wv = tid >> 5;
  int M = lane & 15, hi = lane >> 4;
  for (int tile = wv; tile < 25; tile += 8) {
    int mt = tile / 5, nt = tile % 5;
    V16H a, b; V8F c;
#pragma unroll
    for (int i = 0; i < 16; ++i) a.e[i] = lA[mt * 16 + M][16 * hi + i];
#pragma unroll
    for (int i = 0; i < 16; ++i) b.e[i] = lD[nt * 16 + M][16 * hi + i];
#pragma unroll
    for (int r = 0; r < 8; ++r) c.e[r] = 0.0f;
    c.v = __builtin_amdgcn_wmma_f32_16x16x32_f16(false, a.v, false, b.v,
                                                 (short)0, c.v, false, false);
#pragma unroll
    for (int r = 0; r < 8; ++r) lO[mt * 16 + r + 8 * hi][nt * 16 + M] = c.e[r];
  }

  if (tid < C_) {                       // s[o] = sum_h vmax[h]*WB[o,h]
    float s = 0.0f;
    for (int h = 0; h < H_; ++h) s += sV[h] * (float)lB[tid][h];
    sS[tid] = s;
  }
  __syncthreads();

  for (int t = tid; t < C_ * D_; t += blockDim.x) {
    int o = t % C_, d = t / C_;
    float r = radar[((size_t)n * D_ + d) * W_ + w];
    lO[o][d] += sS[o] * r + b_agg[o];
  }
  __syncthreads();

  int bb = n / N_;
  for (int t = tid; t < D_ * CT_; t += blockDim.x) {
    int d = t / CT_, cch = t % CT_;    // cch fastest -> consecutive atomic addrs
    int idx = sIdx[d];
    if (idx < 0) continue;
    float val;
    if (cch < C_) val = lO[cch][d];
    else          val = pts[(((size_t)n * CP_ + (cch - C_)) * D_ + d) * W_ + w];
    atomicAdd(&acc[((size_t)bb * Y_ * X_ + idx) * CT_ + cch], val);
  }
  for (int d = tid; d < D_; d += blockDim.x) {
    int idx = sIdx[d];
    if (idx < 0) continue;
    atomicAdd(&cnt[(size_t)bb * Y_ * X_ + idx], 1.0f);
  }
}

// ---------------------------------------------------------------------------
// Kernel 4: mean + transpose to (B, CT, Y, X)
// ---------------------------------------------------------------------------
__global__ void k_finalize(const float* __restrict__ acc, const float* __restrict__ cnt,
                           float* __restrict__ out) {
  size_t gid = (size_t)blockIdx.x * blockDim.x + threadIdx.x;
  size_t total = (size_t)B_ * CT_ * Y_ * X_;
  if (gid >= total) return;
  int x  = (int)(gid % X_);
  int y  = (int)((gid / X_) % Y_);
  int ct = (int)((gid / ((size_t)X_ * Y_)) % CT_);
  int b  = (int)(gid / ((size_t)X_ * Y_ * CT_));
  int yx = y * X_ + x;
  float c = cnt[(size_t)b * Y_ * X_ + yx];
  out[gid] = acc[((size_t)b * Y_ * X_ + yx) * CT_ + ct] / fmaxf(c, 1.0f);
}

extern "C" void kernel_launch(void* const* d_in, const int* in_sizes, int n_in,
                              void* d_out, int out_size, void* d_ws, size_t ws_size,
                              hipStream_t stream) {
  const float* df    = (const float*)d_in[0];  // depth_feature
  const float* gv    = (const float*)d_in[1];  // geom_valid
  const float* radar = (const float*)d_in[2];  // radar_occupancy
  const float* pts   = (const float*)d_in[3];  // pts_context
  const float* w_agg = (const float*)d_in[4];
  const float* b_agg = (const float*)d_in[5];
  const int*   geom  = (const int*)d_in[6];    // geom_xyz
  float* out = (float*)d_out;

  char* ws = (char*)d_ws;
  float*    acc   = (float*)(ws + OFF_ACC);
  float*    cnt   = (float*)(ws + OFF_CNT);
  _Float16* dv    = (_Float16*)(ws + OFF_DV);
  float*    vmax  = (float*)(ws + OFF_VMAX);
  _Float16* img16 = (_Float16*)(ws + OFF_IMG);
  _Float16* wcat  = (_Float16*)(ws + OFF_WCAT);
  _Float16* wab   = (_Float16*)(ws + OFF_WAB);

  k_init<<<20480, 256, 0, stream>>>(acc, cnt, wcat, w_agg);
  k_softmax<<<BN_ * H_, 256, 0, stream>>>(df, gv, dv, img16, vmax);
  k_project<<<BN_ * NT2_, 256, 0, stream>>>(img16, wcat, wab);
  k_context_scatter<<<BN_ * W_, 256, 0, stream>>>(wab, dv, vmax, radar, b_agg,
                                                  pts, geom, acc, cnt);
  k_finalize<<<(B_ * CT_ * Y_ * X_ + 255) / 256, 256, 0, stream>>>(acc, cnt, out);
}
